// VishwamAIModel_6356551598368
// MI455X (gfx1250) — matrix operand order repaired and checked by reading, not compiled, not run on hardware
//
#include <hip/hip_runtime.h>
#include <hip/hip_bf16.h>

// ---------------------------------------------------------------------------
// Types for CDNA5 WMMA (wave32): 16x16x32 bf16 -> f32 accumulate
// ---------------------------------------------------------------------------
typedef __attribute__((ext_vector_type(16))) __bf16        v16bf;
typedef __attribute__((ext_vector_type(8)))  float         v8f;
typedef __attribute__((ext_vector_type(8)))  unsigned int  v8u;
typedef __attribute__((ext_vector_type(4)))  unsigned int  v4u;

// Optional gfx1250 async global->LDS path (guarded; falls back to plain copies)
#if defined(__gfx1250__) && __has_builtin(__builtin_amdgcn_global_load_async_to_lds_b128) && __has_builtin(__builtin_amdgcn_s_wait_asynccnt)
#define USE_ASYNC_LDS 1
// The builtin expects: (int4 __device__* src, int4 __lds__* dst, imm offset, imm cpol)
typedef int async_v4i __attribute__((__vector_size__(4 * sizeof(int))));
typedef __attribute__((address_space(1))) async_v4i* async_gptr;
typedef __attribute__((address_space(3))) async_v4i* async_lptr;
#define ASYNC_COPY_B128(gsrc, ldst) \
    __builtin_amdgcn_global_load_async_to_lds_b128((async_gptr)(gsrc), (async_lptr)(ldst), 0, 0)
#endif

__device__ __forceinline__ unsigned short f32_to_bf16_bits(float f) {
    unsigned int u = __builtin_bit_cast(unsigned int, f);
    unsigned int r = (u + 0x7FFFu + ((u >> 16) & 1u)) >> 16;   // round-to-nearest-even
    return (unsigned short)r;
}

__device__ __forceinline__ unsigned int pack_bf16x2(float a, float b) {
    return (unsigned int)f32_to_bf16_bits(a) | ((unsigned int)f32_to_bf16_bits(b) << 16);
}

__device__ __forceinline__ v8f wmma_bf16(v16bf a, v16bf b, v8f c) {
    return __builtin_amdgcn_wmma_f32_16x16x32_bf16(false, a, false, b, (short)0, c,
                                                   false, false);
}

// A fragment (16x32 bf16): lane&15 = M row, (lane>>4)*8 = K base.
// Per lane the 16 values are two contiguous 8-half runs -> 2x ds_load_b128.
template<int LDA>
__device__ __forceinline__ v16bf load_frag_a(const unsigned short* As, int row0, int k0,
                                             int lane) {
    const int m  = row0 + (lane & 15);
    const int kb = k0 + ((lane >> 4) << 3);
    const unsigned short* r = As + m * LDA + kb;
    v4u lo = *(const v4u*)(r);
    v4u hi = *(const v4u*)(r + 16);
    v8u p;
    p[0] = lo[0]; p[1] = lo[1]; p[2] = lo[2]; p[3] = lo[3];
    p[4] = hi[0]; p[5] = hi[1]; p[6] = hi[2]; p[7] = hi[3];
    return __builtin_bit_cast(v16bf, p);
}

// B fragment (32x16 bf16) from a K-transposed LDS tile stored [n][k]:
// n = lane&15, K base = (lane>>4)*8 -> two contiguous 8-half runs (k, k+16).
template<int LDK>
__device__ __forceinline__ v16bf load_frag_bt(const unsigned short* Bt, int k0, int n0,
                                              int lane) {
    const int n  = n0 + (lane & 15);
    const int kb = k0 + ((lane >> 4) << 3);
    const unsigned short* r = Bt + n * LDK + kb;
    v4u lo = *(const v4u*)(r);
    v4u hi = *(const v4u*)(r + 16);
    v8u p;
    p[0] = lo[0]; p[1] = lo[1]; p[2] = lo[2]; p[3] = lo[3];
    p[4] = hi[0]; p[5] = hi[1]; p[6] = hi[2]; p[7] = hi[3];
    return __builtin_bit_cast(v16bf, p);
}

// ---------------------------------------------------------------------------
// Tiled GEMM: C[M,N] = scale * A[M,K] @ B[K,N]   (fp32 in; fp32 or bf16 out)
// 128 threads = 4 waves (2x2), tile 64x64, K step 32, bf16 WMMA math.
// ---------------------------------------------------------------------------
__global__ __launch_bounds__(128)
void gemm_bf16_nn(const float* __restrict__ A, const float* __restrict__ B,
                  void* __restrict__ Cv, int M, int N, int K, float scale,
                  int accumulate, int c_bf16) {
    __shared__ unsigned short As[64 * 32];   // [m][k]
    __shared__ unsigned short Bt[64 * 32];   // [n][k] (transposed tile)

    const int t      = threadIdx.x;
    const int lane   = t & 31;
    const int w      = t >> 5;
    const int waveM  = (w >> 1) * 32;
    const int waveN  = (w & 1) * 32;
    const int blockM = blockIdx.x * 64;
    const int blockN = blockIdx.y * 64;
    const bool interior = (blockM + 64 <= M) && (blockN + 64 <= N) &&
                          ((N & 3) == 0) && ((K & 3) == 0);

    v8f acc[2][2] = {};

    for (int k0 = 0; k0 < K; k0 += 32) {
        if (k0 + 32 < K) {
            __builtin_prefetch(&A[(long long)(blockM + (t >> 1)) * K + k0 + 32], 0, 1);
            __builtin_prefetch(&B[(long long)(k0 + 32 + (t >> 3)) * N + blockN], 0, 1);
        }
        if (interior) {
            // A: 64x32 floats = 512 float4, 4 per thread, b128 loads + b64 LDS stores
#pragma unroll
            for (int i = 0; i < 4; ++i) {
                const int e4 = i * 128 + t;
                const int r  = e4 >> 3;          // 8 float4 per 32-float row
                const int c4 = (e4 & 7) * 4;
                const float4 fv = *(const float4*)&A[(long long)(blockM + r) * K + k0 + c4];
                uint2 pk;
                pk.x = pack_bf16x2(fv.x, fv.y);
                pk.y = pack_bf16x2(fv.z, fv.w);
                *(uint2*)&As[r * 32 + c4] = pk;
            }
            // B: 32x64 floats, read [k][n] coalesced, write transposed [n][k]
#pragma unroll
            for (int i = 0; i < 4; ++i) {
                const int e4 = i * 128 + t;
                const int r  = e4 >> 4;          // 16 float4 per 64-float row
                const int c4 = (e4 & 15) * 4;
                const float4 fv = *(const float4*)&B[(long long)(k0 + r) * N + blockN + c4];
                Bt[(c4 + 0) * 32 + r] = f32_to_bf16_bits(fv.x);
                Bt[(c4 + 1) * 32 + r] = f32_to_bf16_bits(fv.y);
                Bt[(c4 + 2) * 32 + r] = f32_to_bf16_bits(fv.z);
                Bt[(c4 + 3) * 32 + r] = f32_to_bf16_bits(fv.w);
            }
        } else {
#pragma unroll 4
            for (int i = 0; i < 16; ++i) {
                const int e = i * 128 + t;
                const int r = e >> 5, c = e & 31;
                const int gr = blockM + r, gc = k0 + c;
                float v = (gr < M && gc < K) ? A[(long long)gr * K + gc] : 0.0f;
                As[r * 32 + c] = f32_to_bf16_bits(v);
            }
#pragma unroll 4
            for (int i = 0; i < 16; ++i) {
                const int e = i * 128 + t;
                const int r = e >> 6, c = e & 63;   // r = k row, c = n col
                const int gr = k0 + r, gc = blockN + c;
                float v = (gr < K && gc < N) ? B[(long long)gr * N + gc] : 0.0f;
                Bt[c * 32 + r] = f32_to_bf16_bits(v);
            }
        }
        __syncthreads();

        v16bf a0 = load_frag_a<32>(As, waveM + 0, 0, lane);
        v16bf a1 = load_frag_a<32>(As, waveM + 16, 0, lane);
        v16bf b0 = load_frag_bt<32>(Bt, 0, waveN + 0, lane);
        v16bf b1 = load_frag_bt<32>(Bt, 0, waveN + 16, lane);
        acc[0][0] = wmma_bf16(a0, b0, acc[0][0]);
        acc[0][1] = wmma_bf16(a0, b1, acc[0][1]);
        acc[1][0] = wmma_bf16(a1, b0, acc[1][0]);
        acc[1][1] = wmma_bf16(a1, b1, acc[1][1]);
        __syncthreads();
    }

    float* Cf = (float*)Cv;
    unsigned short* Ch = (unsigned short*)Cv;
#pragma unroll
    for (int mi = 0; mi < 2; ++mi)
#pragma unroll
        for (int ni = 0; ni < 2; ++ni)
#pragma unroll
            for (int j = 0; j < 8; ++j) {
                const int m = blockM + waveM + mi * 16 + j + ((lane >> 4) << 3);
                const int n = blockN + waveN + ni * 16 + (lane & 15);
                if (m < M && n < N) {
                    const float r = acc[mi][ni][j] * scale;
                    const long long o = (long long)m * N + n;
                    if (c_bf16)           Ch[o] = f32_to_bf16_bits(r);
                    else if (accumulate)  Cf[o] = Cf[o] + r;
                    else                  Cf[o] = r;
                }
            }
}

// ---------------------------------------------------------------------------
// Fused flash attention over bf16 Q/K/V ([B*S, H*KH] row-major, head = column
// slice). One workgroup per (b, h, 64-query block); online softmax; WMMA for
// both Q.K^T and P.V. Output fp32.
// ---------------------------------------------------------------------------
__global__ __launch_bounds__(128)
void flash_attn(const unsigned short* __restrict__ Q,
                const unsigned short* __restrict__ Kg,
                const unsigned short* __restrict__ V,
                float* __restrict__ O, int S, int Hn, int KHn) {
    __shared__ unsigned short Qs[64 * 64];   // [row][d]
    __shared__ unsigned short Ks[64 * 64];   // [key][d]  (acts as K^T tile)
    __shared__ unsigned short Vt[64 * 64];   // [d][key]  (transposed for P.V)
    __shared__ unsigned short Ps[64 * 64];   // [row][key]
    __shared__ float Ss[64 * 64];
    __shared__ float rowm[64], rowl[64], rowc[64];

    const int t     = threadIdx.x;
    const int lane  = t & 31;
    const int w     = t >> 5;
    const int waveM = (w >> 1) * 32;
    const int waveN = (w & 1) * 32;
    const int mb    = blockIdx.x * 64;
    const int h     = blockIdx.y;
    const int b     = blockIdx.z;
    const int D     = Hn * KHn;
    const long long base = (long long)b * S * D + (long long)h * KHn;

    // stage Q block: 4096 halfs, 4x b128 per thread
#ifdef USE_ASYNC_LDS
#pragma unroll
    for (int i = 0; i < 4; ++i) {
        const int e4 = i * 128 + t;
        const int r = e4 >> 3, c8 = (e4 & 7) * 8;
        ASYNC_COPY_B128(&Q[base + (long long)(mb + r) * D + c8], &Qs[r * 64 + c8]);
    }
#else
#pragma unroll
    for (int i = 0; i < 4; ++i) {
        const int e4 = i * 128 + t;
        const int r = e4 >> 3, c8 = (e4 & 7) * 8;
        *(uint4*)&Qs[r * 64 + c8] = *(const uint4*)&Q[base + (long long)(mb + r) * D + c8];
    }
#endif
    if (t < 64) { rowm[t] = -3.0e38f; rowl[t] = 0.0f; }
    v8f oacc[2][2] = {};
    const float sm_scale = rsqrtf((float)KHn);
#ifdef USE_ASYNC_LDS
    __builtin_amdgcn_s_wait_asynccnt(0);
#endif
    __syncthreads();

    for (int kb = 0; kb < S; kb += 64) {
        // stage K tile (raw copy) ...
#ifdef USE_ASYNC_LDS
#pragma unroll
        for (int i = 0; i < 4; ++i) {
            const int e4 = i * 128 + t;
            const int r = e4 >> 3, c8 = (e4 & 7) * 8;
            ASYNC_COPY_B128(&Kg[base + (long long)(kb + r) * D + c8], &Ks[r * 64 + c8]);
        }
#else
#pragma unroll
        for (int i = 0; i < 4; ++i) {
            const int e4 = i * 128 + t;
            const int r = e4 >> 3, c8 = (e4 & 7) * 8;
            *(uint4*)&Ks[r * 64 + c8] =
                *(const uint4*)&Kg[base + (long long)(kb + r) * D + c8];
        }
#endif
        // ... and V tile transposed [d][key]
#pragma unroll
        for (int i = 0; i < 4; ++i) {
            const int e4 = i * 128 + t;
            const int r = e4 >> 3, c8 = (e4 & 7) * 8;     // r = key, c8 = d base
            const uint4 u = *(const uint4*)&V[base + (long long)(kb + r) * D + c8];
            Vt[(c8 + 0) * 64 + r] = (unsigned short)(u.x & 0xffff);
            Vt[(c8 + 1) * 64 + r] = (unsigned short)(u.x >> 16);
            Vt[(c8 + 2) * 64 + r] = (unsigned short)(u.y & 0xffff);
            Vt[(c8 + 3) * 64 + r] = (unsigned short)(u.y >> 16);
            Vt[(c8 + 4) * 64 + r] = (unsigned short)(u.z & 0xffff);
            Vt[(c8 + 5) * 64 + r] = (unsigned short)(u.z >> 16);
            Vt[(c8 + 6) * 64 + r] = (unsigned short)(u.w & 0xffff);
            Vt[(c8 + 7) * 64 + r] = (unsigned short)(u.w >> 16);
        }
#ifdef USE_ASYNC_LDS
        __builtin_amdgcn_s_wait_asynccnt(0);
#endif
        __syncthreads();

        // S quarter = Q @ K^T (contract over d); Ks[key][d] read as B^T
        v8f sacc[2][2] = {};
#pragma unroll
        for (int d0 = 0; d0 < 64; d0 += 32) {
            v16bf a0 = load_frag_a<64>(Qs, waveM + 0, d0, lane);
            v16bf a1 = load_frag_a<64>(Qs, waveM + 16, d0, lane);
            v16bf b0 = load_frag_bt<64>(Ks, d0, waveN + 0, lane);
            v16bf b1 = load_frag_bt<64>(Ks, d0, waveN + 16, lane);
            sacc[0][0] = wmma_bf16(a0, b0, sacc[0][0]);
            sacc[0][1] = wmma_bf16(a0, b1, sacc[0][1]);
            sacc[1][0] = wmma_bf16(a1, b0, sacc[1][0]);
            sacc[1][1] = wmma_bf16(a1, b1, sacc[1][1]);
        }
#pragma unroll
        for (int mi = 0; mi < 2; ++mi)
#pragma unroll
            for (int ni = 0; ni < 2; ++ni)
#pragma unroll
                for (int j = 0; j < 8; ++j) {
                    const int m = waveM + mi * 16 + j + ((lane >> 4) << 3);
                    const int n = waveN + ni * 16 + (lane & 15);
                    Ss[m * 64 + n] = sacc[mi][ni][j] * sm_scale;
                }
        __syncthreads();

        // online softmax per query row
        if (t < 64) {
            float mx = rowm[t];
#pragma unroll 8
            for (int c = 0; c < 64; ++c) mx = fmaxf(mx, Ss[t * 64 + c]);
            const float corr = __expf(rowm[t] - mx);
            float sum = 0.0f;
#pragma unroll 8
            for (int c = 0; c < 64; ++c) {
                const float p = __expf(Ss[t * 64 + c] - mx);
                sum += p;
                Ps[t * 64 + c] = f32_to_bf16_bits(p);
            }
            rowc[t] = corr;
            rowm[t] = mx;
            rowl[t] = rowl[t] * corr + sum;
        }
        __syncthreads();

        // rescale O accumulators, then O += P @ V (contract over keys)
#pragma unroll
        for (int mi = 0; mi < 2; ++mi)
#pragma unroll
            for (int j = 0; j < 8; ++j) {
                const int m = waveM + mi * 16 + j + ((lane >> 4) << 3);
                const float c = rowc[m];
                oacc[mi][0][j] *= c;
                oacc[mi][1][j] *= c;
            }
#pragma unroll
        for (int kk = 0; kk < 64; kk += 32) {
            v16bf a0 = load_frag_a<64>(Ps, waveM + 0, kk, lane);
            v16bf a1 = load_frag_a<64>(Ps, waveM + 16, kk, lane);
            v16bf b0 = load_frag_bt<64>(Vt, kk, waveN + 0, lane);
            v16bf b1 = load_frag_bt<64>(Vt, kk, waveN + 16, lane);
            oacc[0][0] = wmma_bf16(a0, b0, oacc[0][0]);
            oacc[0][1] = wmma_bf16(a0, b1, oacc[0][1]);
            oacc[1][0] = wmma_bf16(a1, b0, oacc[1][0]);
            oacc[1][1] = wmma_bf16(a1, b1, oacc[1][1]);
        }
        __syncthreads();
    }

#pragma unroll
    for (int mi = 0; mi < 2; ++mi)
#pragma unroll
        for (int ni = 0; ni < 2; ++ni)
#pragma unroll
            for (int j = 0; j < 8; ++j) {
                const int m = waveM + mi * 16 + j + ((lane >> 4) << 3);
                const int n = waveN + ni * 16 + (lane & 15);
                const float inv = 1.0f / rowl[m];
                O[base + (long long)(mb + m) * D + n] = oacc[mi][ni][j] * inv;
            }
}

// ---------------------------------------------------------------------------
// Elementwise helpers (float4 vectorized; D % 4 == 0)
// ---------------------------------------------------------------------------
__global__ void gather_emb(const int* __restrict__ tokens, const float* __restrict__ emb,
                           float* __restrict__ x, int n, int D) {
    long long t4 = (long long)blockIdx.x * blockDim.x + threadIdx.x;
    const int d4 = D >> 2;
    if (t4 >= (long long)n * d4) return;
    const int row = (int)(t4 / d4);
    const int c4  = (int)(t4 % d4) * 4;
    *(float4*)&x[(long long)row * D + c4] =
        *(const float4*)&emb[(long long)tokens[row] * D + c4];
}

__global__ void gate_argmax(const float* __restrict__ gate, int* __restrict__ idx,
                            int n, int E) {
    int t = blockIdx.x * blockDim.x + threadIdx.x;
    if (t >= n) return;
    const float* g = gate + (long long)t * E;
    int best = 0;
    float bv = g[0];
    for (int e = 1; e < E; ++e) {
        float v = g[e];
        if (v > bv) { bv = v; best = e; }
    }
    idx[t] = best;
}

__global__ void mask_copy(const float* __restrict__ h, const int* __restrict__ idx,
                          float* __restrict__ xe, int n, int D, int e) {
    long long t4 = (long long)blockIdx.x * blockDim.x + threadIdx.x;
    const int d4 = D >> 2;
    if (t4 >= (long long)n * d4) return;
    const int row = (int)(t4 / d4);
    const long long o = t4 * 4;
    if (idx[row] == e) {
        *(float4*)&xe[o] = *(const float4*)&h[o];
    } else {
        *(float4*)&xe[o] = make_float4(0.f, 0.f, 0.f, 0.f);
    }
}

__global__ __launch_bounds__(256)
void out_proj(const float* __restrict__ hf, const float* __restrict__ Wout,
              float* __restrict__ out, int n, int D) {
    int t = blockIdx.x * blockDim.x + threadIdx.x;
    if (t >= n) return;
    const float* r = hf + (long long)t * D;
    float s = 0.0f;
#pragma unroll 4
    for (int c = 0; c < D; c += 4) {
        const float4 a = *(const float4*)&r[c];
        const float4 b = *(const float4*)&Wout[c];
        s += a.x * b.x + a.y * b.y + a.z * b.z + a.w * b.w;
    }
    out[t] = s;
}

// ---------------------------------------------------------------------------
// Host orchestration
// ---------------------------------------------------------------------------
extern "C" void kernel_launch(void* const* d_in, const int* in_sizes, int n_in,
                              void* d_out, int out_size, void* d_ws, size_t ws_size,
                              hipStream_t stream) {
    (void)in_sizes; (void)n_in; (void)out_size; (void)ws_size;

    const int* tokens = (const int*)d_in[0];
    const float* emb  = (const float*)d_in[1];
    const float* Wq   = (const float*)d_in[2];
    const float* Wk   = (const float*)d_in[3];
    const float* Wv   = (const float*)d_in[4];
    const float* Wo   = (const float*)d_in[5];
    const float* W1   = (const float*)d_in[6];
    const float* W2   = (const float*)d_in[7];
    const float* Wg   = (const float*)d_in[8];
    const float* eWq  = (const float*)d_in[9];
    const float* eWk  = (const float*)d_in[10];
    const float* eWv  = (const float*)d_in[11];
    const float* eWo  = (const float*)d_in[12];
    const float* eW1  = (const float*)d_in[13];
    const float* eW2  = (const float*)d_in[14];
    const float* Wout = (const float*)d_in[15];

    const int B = 8, S = 1024, D = 512, Hh = 8, KH = 64, FF = 2048, E = 4;
    const int Nt = B * S;
    const size_t C512 = (size_t)Nt * D;

    float* ws   = (float*)d_ws;
    float* x    = ws;                            // embeddings / masked xe (fp32)
    float* qh   = x + C512;                      // bf16 q (uses half the slot)
    float* kh   = qh + C512;                     // bf16 k
    float* vh   = kh + C512;                     // bf16 v
    float* ao   = vh + C512;                     // attention output (fp32)
    float* t0   = ao + C512;                     // attn @ Wo
    float* hbuf = t0 + C512;                     // shared block output h
    float* hout = hbuf + C512;                   // expert sum accumulator
    float* f1   = hout + C512;                   // [Nt, FF]
    float* gate = f1 + (size_t)Nt * FF;          // [Nt, E]
    int*   idx  = (int*)(gate + (size_t)Nt * E);

    dim3 blk(128);
    auto gemm = [&](const float* A, const float* Bm, void* C, int M, int N, int Kd,
                    float sc, int accum, int cbf16) {
        dim3 grid((M + 63) / 64, (N + 63) / 64);
        gemm_bf16_nn<<<grid, blk, 0, stream>>>(A, Bm, C, M, N, Kd, sc, accum, cbf16);
    };

    auto transformer_block = [&](const float* xin, const float* wq, const float* wk,
                                 const float* wv, const float* wo, const float* w1,
                                 const float* w2, float* outBuf, int accum) {
        gemm(xin, wq, qh, Nt, D, D, 1.0f, 0, 1);   // bf16 outputs for attention
        gemm(xin, wk, kh, Nt, D, D, 1.0f, 0, 1);
        gemm(xin, wv, vh, Nt, D, D, 1.0f, 0, 1);
        dim3 fg(S / 64, Hh, B);
        flash_attn<<<fg, blk, 0, stream>>>((const unsigned short*)qh,
                                           (const unsigned short*)kh,
                                           (const unsigned short*)vh, ao, S, Hh, KH);
        gemm(ao, wo, t0, Nt, D, D, 1.0f, 0, 0);
        gemm(t0, w1, f1, Nt, FF, D, 1.0f, 0, 0);
        gemm(f1, w2, outBuf, Nt, D, FF, 1.0f, accum, 0);
    };

    // 1. embedding gather
    {
        long long tot = (long long)Nt * (D / 4);
        gather_emb<<<(unsigned)((tot + 255) / 256), 256, 0, stream>>>(tokens, emb, x, Nt, D);
    }

    // 2. shared transformer block -> hbuf
    transformer_block(x, Wq, Wk, Wv, Wo, W1, W2, hbuf, 0);

    // 3. gating + argmax
    gemm(hbuf, Wg, gate, Nt, E, D, 1.0f, 0, 0);
    gate_argmax<<<(Nt + 255) / 256, 256, 0, stream>>>(gate, idx, Nt, E);

    // 4. experts: masked dense dispatch, accumulate into hout
    for (int e = 0; e < E; ++e) {
        long long tot = (long long)Nt * (D / 4);
        mask_copy<<<(unsigned)((tot + 255) / 256), 256, 0, stream>>>(hbuf, idx, x, Nt, D, e);
        transformer_block(x,
                          eWq + (size_t)e * D * D,
                          eWk + (size_t)e * D * D,
                          eWv + (size_t)e * D * D,
                          eWo + (size_t)e * D * D,
                          eW1 + (size_t)e * D * FF,
                          eW2 + (size_t)e * FF * D,
                          hout, e > 0);
    }

    // 5. final projection to [B,S,1]
    out_proj<<<(Nt + 255) / 256, 256, 0, stream>>>(hout, Wout, (float*)d_out, Nt, D);
}
